// mLSTMLayerV1_48541720379869
// MI455X (gfx1250) — matrix operand order repaired and checked by reading, not compiled, not run on hardware
//
#include <hip/hip_runtime.h>

// ---------------------------------------------------------------------------
// mLSTM layer (xLSTM parallel form) for gfx1250 (MI455X).
// f16 data path, f32 WMMA accumulation, async global->LDS staging.
// ---------------------------------------------------------------------------

typedef __attribute__((ext_vector_type(16))) _Float16 v16h;
typedef __attribute__((ext_vector_type(8)))  _Float16 v8h;
typedef __attribute__((ext_vector_type(8)))  float    v8f;

#define E_DIM 1024
#define NH    4
#define DH    256
#define SEQ   2048
#define GATE_CAP 15.0f
#define NEG_BIG (-1.0e30f)

__device__ __forceinline__ float soft_cap_f(float v) {
    return GATE_CAP * tanhf(v * (1.0f / GATE_CAP));
}
__device__ __forceinline__ float log_sigmoid_f(float v) {
    return fminf(v, 0.0f) - log1pf(expf(-fabsf(v)));
}

// ---------------------------------------------------------------------------
// fp32 -> f16 bulk conversion (vectorized, 8 elems/thread)
// ---------------------------------------------------------------------------
__global__ __launch_bounds__(256)
void cvt_f32_f16(const float* __restrict__ in, _Float16* __restrict__ out, int n) {
    int i = (blockIdx.x * 256 + threadIdx.x) * 8;
    if (i >= n) return;
    v8h o;
    #pragma unroll
    for (int e = 0; e < 8; ++e) o[e] = (_Float16)in[i + e];
    *(v8h*)(out + i) = o;
}

// ---------------------------------------------------------------------------
// C = A @ B, A/B f16 row-major, OutT in {float, _Float16}.
// Tile 64x64, BK=32, 128 threads (4 waves, one 64x16 strip each).
// A tile staged via async global->LDS b128 (ASYNCcnt); B tile staged
// transposed via vector loads + ds scatter.
// ---------------------------------------------------------------------------
#define BM 64
#define BN 64
#define BK 32

template <typename OutT>
__global__ __launch_bounds__(128)
void gemm_wmma_h(const _Float16* __restrict__ A, const _Float16* __restrict__ B,
                 OutT* __restrict__ C, int M, int N, int K) {
    __shared__ __align__(32) _Float16 Ah[BM * BK];  // [row][k] == memory layout
    __shared__ __align__(32) _Float16 Bt[BN * BK];  // [n][k]   (transposed)

    const int tid  = threadIdx.x;
    const int wave = tid >> 5;
    const int lane = tid & 31;
    const int l    = lane & 15;
    const int hi   = lane >> 4;

    const int m0 = blockIdx.y * BM;
    const int n0 = blockIdx.x * BN;

    // LDS byte offset of Ah (flat shared address truncates to LDS offset)
    const unsigned ahBase = (unsigned)(unsigned long long)(&Ah[0]);

    v8f acc[4] = {};

    for (int k0 = 0; k0 < K; k0 += BK) {
        // ---- A tile: 64x32 f16 = 4KB = 256 x 16B chunks, async to LDS ----
        #pragma unroll
        for (int c2 = 0; c2 < 2; ++c2) {
            int c = tid + c2 * 128;                 // chunk id 0..255
            unsigned ldsoff = ahBase + c * 16;
            const _Float16* gp =
                A + (size_t)(m0 + (c >> 2)) * K + k0 + (c & 3) * 8;
            asm volatile("global_load_async_to_lds_b128 %0, %1, off"
                         :: "v"(ldsoff), "v"(gp) : "memory");
        }

        // ---- B tile: 32x64 -> Bt[n][k]; vector load + ds scatter ----
        #pragma unroll
        for (int c2 = 0; c2 < 2; ++c2) {
            int c  = tid + c2 * 128;                // 256 chunks of 8 halves
            int kk = c >> 3;
            int nn = (c & 7) * 8;
            v8h bv = *(const v8h*)(B + (size_t)(k0 + kk) * N + n0 + nn);
            #pragma unroll
            for (int e = 0; e < 8; ++e) Bt[(nn + e) * BK + kk] = bv[e];
        }

        if (k0 + BK < K)  // prefetch next B tile row region
            __builtin_prefetch(B + (size_t)(k0 + BK) * N + n0 + (tid & 63), 0, 1);

        asm volatile("s_wait_asynccnt 0x0" ::: "memory");
        __syncthreads();

        // B fragment (32x16): lane n<16 -> col n, K 0..15; lane>=16 -> K 16..31
        const _Float16* bp = &Bt[(wave * 16 + l) * BK + hi * 16];
        v16h bf = *(const v16h*)bp;

        #pragma unroll
        for (int r = 0; r < 4; ++r) {
            // A fragment (16x32): lane -> row lane&15; K contiguous 16*hi..16*hi+15
            const _Float16* ap = &Ah[(r * 16 + l) * BK];
            v8h lo  = *(const v8h*)(ap + hi * 16);
            v8h hiv = *(const v8h*)(ap + hi * 16 + 8);
            v16h af = __builtin_shufflevector(lo, hiv,
                        0,1,2,3,4,5,6,7,8,9,10,11,12,13,14,15);
            acc[r] = __builtin_amdgcn_wmma_f32_16x16x32_f16(
                         false, af, false, bf, (short)0, acc[r], false, false);
        }
        __syncthreads();
    }

    #pragma unroll
    for (int r = 0; r < 4; ++r) {
        #pragma unroll
        for (int j = 0; j < 8; ++j) {
            int row = m0 + r * 16 + hi * 8 + j;
            int col = n0 + wave * 16 + l;
            C[(size_t)row * N + col] = (OutT)acc[r][j];
        }
    }
}

// ---------------------------------------------------------------------------
// Gate projections (E -> NH): ig = softcap(x@Wig+big); fg -> logsigmoid.
// ---------------------------------------------------------------------------
__global__ __launch_bounds__(128)
void gates_kernel(const float* __restrict__ x,
                  const float* __restrict__ Wig, const float* __restrict__ big,
                  const float* __restrict__ Wfg, const float* __restrict__ bfg,
                  float* __restrict__ ig_sc, float* __restrict__ logfg) {
    const int row = blockIdx.x;
    const int tid = threadIdx.x;
    const float* xr = x + (size_t)row * E_DIM;

    float acc[8] = {};
    for (int k = tid; k < E_DIM; k += 128) {
        float xv = xr[k];
        #pragma unroll
        for (int h = 0; h < NH; ++h) {
            acc[h]     += xv * Wig[k * NH + h];
            acc[4 + h] += xv * Wfg[k * NH + h];
        }
    }
    __shared__ float red[128 * 8];
    #pragma unroll
    for (int j = 0; j < 8; ++j) red[tid * 8 + j] = acc[j];
    __syncthreads();
    for (int s = 64; s > 0; s >>= 1) {
        if (tid < s) {
            #pragma unroll
            for (int j = 0; j < 8; ++j) red[tid * 8 + j] += red[(tid + s) * 8 + j];
        }
        __syncthreads();
    }
    if (tid < NH) {
        int h = tid;
        float igv = soft_cap_f(red[h] + big[h]);
        float fgv = soft_cap_f(red[4 + h] + bfg[h]);
        ig_sc[(size_t)row * NH + h] = igv;
        logfg[(size_t)row * NH + h] = log_sigmoid_f(fgv);
    }
}

// ---------------------------------------------------------------------------
// Per-(b,h) inclusive scan of log-forget over S + ig transpose to [B*NH,S].
// ---------------------------------------------------------------------------
__global__ __launch_bounds__(1024)
void scan_kernel(const float* __restrict__ logfg, const float* __restrict__ ig_sc,
                 float* __restrict__ lfcum, float* __restrict__ igT) {
    const int bh = blockIdx.x;
    const int b  = bh >> 2;       // NH == 4
    const int h  = bh & 3;
    const int t  = threadIdx.x;

    const int s0 = 2 * t, s1 = 2 * t + 1;
    float a0 = logfg[((size_t)b * SEQ + s0) * NH + h];
    float a1 = logfg[((size_t)b * SEQ + s1) * NH + h];
    float l0 = a0;
    float l1 = a0 + a1;

    __shared__ float sums[1024];
    sums[t] = l1;
    __syncthreads();
    for (int off = 1; off < 1024; off <<= 1) {
        float add = (t >= off) ? sums[t - off] : 0.0f;
        __syncthreads();
        sums[t] += add;
        __syncthreads();
    }
    float ex = (t > 0) ? sums[t - 1] : 0.0f;

    lfcum[(size_t)bh * SEQ + s0] = ex + l0;
    lfcum[(size_t)bh * SEQ + s1] = ex + l1;
    igT[(size_t)bh * SEQ + s0] = ig_sc[((size_t)b * SEQ + s0) * NH + h];
    igT[(size_t)bh * SEQ + s1] = ig_sc[((size_t)b * SEQ + s1) * NH + h];
}

// ---------------------------------------------------------------------------
// Flash-style stabilized parallel mLSTM, f16 q/k/v inputs.
// Grid: (S/16, B*NH); 1 wave per 16-row tile. Per 32-col block:
//   V block staged transposed in LDS; scores via WMMA (direct aligned global
//   f16 fragment loads); gate/mask/online-max; C->LDS transpose; h += C@V.
// ---------------------------------------------------------------------------
__global__ __launch_bounds__(32)
void mlstm_kernel(const _Float16* __restrict__ q, const _Float16* __restrict__ k,
                  const _Float16* __restrict__ v, const float* __restrict__ lfcum,
                  const float* __restrict__ igT, float* __restrict__ hbuf) {
    const int it = blockIdx.x;          // row-tile index
    const int bh = blockIdx.y;
    const int b  = bh >> 2;
    const int h  = bh & 3;
    const int i0 = it * 16;

    const int lane = threadIdx.x;
    const int l    = lane & 15;
    const int hi   = lane >> 4;

    const _Float16* qbh = q + (size_t)b * SEQ * E_DIM + h * DH;
    const _Float16* kbh = k + (size_t)b * SEQ * E_DIM + h * DH;
    const _Float16* vbh = v + (size_t)b * SEQ * E_DIM + h * DH;
    const float* lfc = lfcum + (size_t)bh * SEQ;
    const float* igv = igT   + (size_t)bh * SEQ;

    __shared__ __align__(32) _Float16 Cbuf[16 * 32];   // C tile (D->A relayout)
    __shared__ __align__(32) _Float16 Vt[DH * 32];     // [d][j] transposed V blk

    // ---- preload Q A-fragments (8 slices of K=32), aligned b128 loads ----
    v16h qf[8];
    #pragma unroll
    for (int db = 0; db < 8; ++db) {
        const _Float16* qr = qbh + (size_t)(i0 + l) * E_DIM + db * 32;
        v8h lo  = *(const v8h*)(qr + hi * 16);
        v8h hiv = *(const v8h*)(qr + hi * 16 + 8);
        qf[db] = __builtin_shufflevector(lo, hiv,
                     0,1,2,3,4,5,6,7,8,9,10,11,12,13,14,15);
    }

    v8f hacc[16];
    #pragma unroll
    for (int t = 0; t < 16; ++t) hacc[t] = (v8f){};
    float rowsum[8], mrow[8], lfci[8];
    #pragma unroll
    for (int j = 0; j < 8; ++j) {
        rowsum[j] = 0.0f;
        mrow[j]   = NEG_BIG;
        lfci[j]   = lfc[i0 + hi * 8 + j];
    }

    const float SCALE = 0.0625f;        // 1/sqrt(DH)
    const int nblk = (it >> 1) + 1;

    for (int blk = 0; blk < nblk; ++blk) {
        const int j0 = blk * 32;

        // ---- stage V block transposed: Vt[d][j], lane covers 8 d-columns ----
        for (int jj = 0; jj < 32; ++jj) {
            v8h vr = *(const v8h*)(vbh + (size_t)(j0 + jj) * E_DIM + lane * 8);
            #pragma unroll
            for (int e = 0; e < 8; ++e) Vt[(lane * 8 + e) * 32 + jj] = vr[e];
        }

        // ---- scores: s0 (cols j0..+15), s1 (cols j0+16..+31) ----
        v8f s0 = {}, s1 = {};
        #pragma unroll
        for (int db = 0; db < 8; ++db) {
            // B = K^T fragment: lane -> col j, halves -> d (contiguous 32B)
            v16h kf0 = *(const v16h*)(kbh + (size_t)(j0 + l) * E_DIM
                                          + db * 32 + hi * 16);
            v16h kf1 = *(const v16h*)(kbh + (size_t)(j0 + 16 + l) * E_DIM
                                          + db * 32 + hi * 16);
            s0 = __builtin_amdgcn_wmma_f32_16x16x32_f16(false, qf[db], false, kf0,
                                                        (short)0, s0, false, false);
            s1 = __builtin_amdgcn_wmma_f32_16x16x32_f16(false, qf[db], false, kf1,
                                                        (short)0, s1, false, false);
        }

        // ---- gating + online stabilization ----
        const int jA = j0 + l;
        const int jB = jA + 16;
        const float cv0 = igv[jA] - lfc[jA];
        const float cv1 = igv[jB] - lfc[jB];

        __syncthreads();
        #pragma unroll
        for (int jj = 0; jj < 8; ++jj) {
            const int row = i0 + hi * 8 + jj;
            float ld0 = (row >= jA) ? (lfci[jj] + cv0) : NEG_BIG;
            float ld1 = (row >= jB) ? (lfci[jj] + cv1) : NEG_BIG;
            float tm = fmaxf(ld0, ld1);
            #pragma unroll
            for (int msk = 1; msk < 16; msk <<= 1)
                tm = fmaxf(tm, __shfl_xor(tm, msk));   // 16-lane row max
            float nm = fmaxf(mrow[jj], tm);
            float sc = expf(mrow[jj] - nm);            // first iter -> 0
            mrow[jj] = nm;
            rowsum[jj] *= sc;
            #pragma unroll
            for (int t = 0; t < 16; ++t) hacc[t][jj] *= sc;

            float c0 = (row >= jA) ? s0[jj] * SCALE * expf(ld0 - nm) : 0.0f;
            float c1 = (row >= jB) ? s1[jj] * SCALE * expf(ld1 - nm) : 0.0f;
            rowsum[jj] += c0 + c1;
            Cbuf[(hi * 8 + jj) * 32 + l]      = (_Float16)c0;
            Cbuf[(hi * 8 + jj) * 32 + 16 + l] = (_Float16)c1;
        }
        __syncthreads();

        // ---- reload C as WMMA A-fragment (16 x K=32 over j) ----
        const _Float16* cp = &Cbuf[l * 32];
        v8h clo = *(const v8h*)(cp + hi * 16);
        v8h chi = *(const v8h*)(cp + hi * 16 + 8);
        v16h cf = __builtin_shufflevector(clo, chi,
                      0,1,2,3,4,5,6,7,8,9,10,11,12,13,14,15);

        // ---- h += C @ V over 16 DH tiles; V frags straight from Vt ----
        #pragma unroll
        for (int dt = 0; dt < 16; ++dt) {
            v16h vf = *(const v16h*)(&Vt[(dt * 16 + l) * 32 + hi * 16]);
            hacc[dt] = __builtin_amdgcn_wmma_f32_16x16x32_f16(
                           false, cf, false, vf, (short)0, hacc[dt], false, false);
        }
        __syncthreads();
    }

    // ---- finalize: n = max(|sum C|, exp(-m)); h /= (n + eps) ----
    float inv[8];
    #pragma unroll
    for (int jj = 0; jj < 8; ++jj) {
        float rs = rowsum[jj];
        #pragma unroll
        for (int msk = 1; msk < 16; msk <<= 1) rs += __shfl_xor(rs, msk);
        float n = fmaxf(fabsf(rs), expf(-mrow[jj]));
        inv[jj] = 1.0f / (n + 1.0e-6f);
    }
    #pragma unroll
    for (int dt = 0; dt < 16; ++dt) {
        #pragma unroll
        for (int jj = 0; jj < 8; ++jj) {
            int row = i0 + hi * 8 + jj;
            hbuf[((size_t)b * SEQ + row) * E_DIM + h * DH + dt * 16 + l] =
                hacc[dt][jj] * inv[jj];
        }
    }
}

// ---------------------------------------------------------------------------
// Per-head LayerNorm (over DH) * norm_w, gated by sigmoid(o); f16 output.
// ---------------------------------------------------------------------------
__global__ __launch_bounds__(1024)
void norm_gate_kernel(const float* __restrict__ hbuf, const float* __restrict__ obuf,
                      const float* __restrict__ normw, _Float16* __restrict__ hout) {
    const int row = blockIdx.x;
    const int tid = threadIdx.x;
    const int d   = tid & 255;
    const int base = tid & ~255;

    float val = hbuf[(size_t)row * E_DIM + tid];
    __shared__ float red[1024];
    red[tid] = val;
    __syncthreads();
    for (int s = 128; s > 0; s >>= 1) {
        if (d < s) red[tid] += red[tid + s];
        __syncthreads();
    }
    float mu = red[base] * (1.0f / 256.0f);
    __syncthreads();
    float df = val - mu;
    red[tid] = df * df;
    __syncthreads();
    for (int s = 128; s > 0; s >>= 1) {
        if (d < s) red[tid] += red[tid + s];
        __syncthreads();
    }
    float var = red[base] * (1.0f / 256.0f);
    float hn = df * rsqrtf(var + 1.0e-6f) * normw[tid];
    float ov = obuf[(size_t)row * E_DIM + tid];
    hout[(size_t)row * E_DIM + tid] = (_Float16)(hn / (1.0f + expf(-ov)));
}

// ---------------------------------------------------------------------------
// kernel_launch
// Inputs: 0:x 1:Wq 2:Wk 3:Wv 4:Wo 5:Wig 6:big 7:Wfg 8:bfg 9:norm_w 10:Wdown
// ---------------------------------------------------------------------------
extern "C" void kernel_launch(void* const* d_in, const int* in_sizes, int n_in,
                              void* d_out, int out_size, void* d_ws, size_t ws_size,
                              hipStream_t stream) {
    const float* x     = (const float*)d_in[0];
    const float* Wq    = (const float*)d_in[1];
    const float* Wk    = (const float*)d_in[2];
    const float* Wv    = (const float*)d_in[3];
    const float* Wo    = (const float*)d_in[4];
    const float* Wig   = (const float*)d_in[5];
    const float* big   = (const float*)d_in[6];
    const float* Wfg   = (const float*)d_in[7];
    const float* bfg   = (const float*)d_in[8];
    const float* normw = (const float*)d_in[9];
    const float* Wdown = (const float*)d_in[10];
    float* out = (float*)d_out;

    const int BS  = in_sizes[0] / E_DIM;   // B * S (= 4096)
    const int Bn  = BS / SEQ;              // B (= 2)
    const int NT  = BS * E_DIM;            // 4M elements
    const int WN  = E_DIM * E_DIM;         // 1M elements per big weight
    const int GN  = BS * NH;

    // ---- workspace layout ----
    float* ob   = (float*)d_ws;                 // [BS,E] o-projection (f32)
    float* hb   = ob + (size_t)NT;              // [BS,E] mlstm output (f32)
    float* gsc  = hb + (size_t)NT;              // ig softcapped [BS,NH]
    float* glog = gsc + (size_t)GN;             // logsigmoid(fg) [BS,NH]
    float* lfcu = glog + (size_t)GN;            // [B*NH,S]
    float* igT  = lfcu + (size_t)Bn * NH * SEQ; // [B*NH,S]
    _Float16* x16   = (_Float16*)(igT + (size_t)Bn * NH * SEQ);
    _Float16* q16   = x16 + (size_t)NT;
    _Float16* k16   = q16 + (size_t)NT;
    _Float16* v16   = k16 + (size_t)NT;
    _Float16* wcvt  = v16 + (size_t)NT;         // reused per weight
    _Float16* hout16 = q16;                     // alias: q dead after mlstm

    dim3 gemm_grid(E_DIM / BN, BS / BM);        // (16, 64)
    const int cvtN_x = NT / 2048, cvtN_w = WN / 2048;

    cvt_f32_f16<<<cvtN_x, 256, 0, stream>>>(x, x16, NT);

    cvt_f32_f16<<<cvtN_w, 256, 0, stream>>>(Wq, wcvt, WN);
    gemm_wmma_h<_Float16><<<gemm_grid, 128, 0, stream>>>(x16, wcvt, q16, BS, E_DIM, E_DIM);
    cvt_f32_f16<<<cvtN_w, 256, 0, stream>>>(Wk, wcvt, WN);
    gemm_wmma_h<_Float16><<<gemm_grid, 128, 0, stream>>>(x16, wcvt, k16, BS, E_DIM, E_DIM);
    cvt_f32_f16<<<cvtN_w, 256, 0, stream>>>(Wv, wcvt, WN);
    gemm_wmma_h<_Float16><<<gemm_grid, 128, 0, stream>>>(x16, wcvt, v16, BS, E_DIM, E_DIM);
    cvt_f32_f16<<<cvtN_w, 256, 0, stream>>>(Wo, wcvt, WN);
    gemm_wmma_h<float><<<gemm_grid, 128, 0, stream>>>(x16, wcvt, ob, BS, E_DIM, E_DIM);

    gates_kernel<<<BS, 128, 0, stream>>>(x, Wig, big, Wfg, bfg, gsc, glog);
    scan_kernel<<<Bn * NH, 1024, 0, stream>>>(glog, gsc, lfcu, igT);

    mlstm_kernel<<<dim3(SEQ / 16, Bn * NH), 32, 0, stream>>>(q16, k16, v16,
                                                             lfcu, igT, hb);

    norm_gate_kernel<<<BS, 1024, 0, stream>>>(hb, ob, normw, hout16);

    cvt_f32_f16<<<cvtN_w, 256, 0, stream>>>(Wdown, wcvt, WN);
    gemm_wmma_h<float><<<gemm_grid, 128, 0, stream>>>(hout16, wcvt, out, BS, E_DIM, E_DIM);
}
